// GraphFeatureExtractor_44349832298690
// MI455X (gfx1250) — matrix operand order repaired and checked by reading, not compiled, not run on hardware
//
#include <hip/hip_runtime.h>
#include <hip/hip_bf16.h>

// ---------------------------------------------------------------------------
// HGT graph feature extractor for MI455X (gfx1250, wave32, WMMA).
// GEMMs: v_wmma_f32_16x16x32_f16, A tile staged to LDS via async global->LDS
// (ASYNCcnt, double buffered), B streamed coalesced from L2-resident weights.
// ---------------------------------------------------------------------------

typedef __attribute__((ext_vector_type(16))) _Float16 v16h;
typedef __attribute__((ext_vector_type(8)))  float    v8f;
typedef int v4i __attribute__((__vector_size__(16)));   // matches builtin param

#define D_OUT 4096
#define HEADS 8
#define HD    512
#define KSTEP 64

#if __has_builtin(__builtin_amdgcn_global_load_async_to_lds_b128) && \
    __has_builtin(__builtin_amdgcn_s_wait_asynccnt)
#define USE_ASYNC_LDS 1
#else
#define USE_ASYNC_LDS 0
#endif

// ---------------------------------------------------------------------------
// Generic (optionally batched) GEMM:  C = A[MxK] * B[KxN] (+bias) (+rowAdd)
// Row-major f32 in memory, f16 WMMA compute, f32 accumulate.
// Block = 256 threads = 8 waves; each wave owns a 16x16 C tile; block tile is
// 16(M) x 128(N); N must be a multiple of 128 (true at every call site).
// A tile (16 x 64 f32, shared by all 8 waves) is staged into LDS with
// double-buffered async global->LDS copies (one b128 per thread per step).
// B addresses: uniform per-iteration base (1 scalar mul / iter) + per-lane
// 32-bit invariant offset + hoisted j*ldb constants -> 1 add per load.
// WMMA lane layouts per CDNA5 ISA 7.12.2:
//   A (16x32 f16): m = lane&15; elems 0..7 -> K = half*8 + j,
//                               elems 8..15 -> K = 16 + half*8 + j
//   B (32x16 f16): n = lane&15; elems 0..7 -> K = half*8 + j, elems 8..15 -> K = 16 + half*8 + j
//   D (16x16 f32): n = lane&15; vgpr r -> M = r + 8*half
// ---------------------------------------------------------------------------
__global__ __launch_bounds__(256)
void gemm_wmma_f16(const float* __restrict__ A, int lda, long long sA,
                   const float* __restrict__ B, int ldb, long long sB,
                   float* __restrict__ C, int ldc, long long sC,
                   const float* __restrict__ bias,
                   const float* __restrict__ rowAdd, int ldRowAdd,
                   int M, int N, int K)
{
    __shared__ float tileA[2][16 * KSTEP];   // 2 x 4 KB double buffer

    const int lane = threadIdx.x & 31;
    const int wave = threadIdx.x >> 5;
    const int half = lane >> 4;       // 0: lanes 0-15, 1: lanes 16-31
    const int mn   = lane & 15;

    const int col0 = blockIdx.x * 128 + wave * 16;
    const int row0 = blockIdx.y * 16;

    A += (long long)blockIdx.z * sA;
    B += (long long)blockIdx.z * sB;
    C += (long long)blockIdx.z * sC;

    // --- A-tile copy assignment: thread t moves 16B (row t>>4, 4 floats) ---
    const int tcRow = threadIdx.x >> 4;          // 0..15
    const int tcCol = (threadIdx.x & 15) * 4;    // 0,4,...,60
    int rowACopy = row0 + tcRow;                 // clamp edge rows (M=300 case)
    if (rowACopy > M - 1) rowACopy = M - 1;
    const float* __restrict__ Acopy = A + (long long)rowACopy * lda + tcCol;
    float* __restrict__ ldsDst0 = &tileA[0][tcRow * KSTEP + tcCol];
    float* __restrict__ ldsDst1 = &tileA[1][tcRow * KSTEP + tcCol];

#if USE_ASYNC_LDS
#define ISSUE_COPY(bufp, kkv)                                                  \
    __builtin_amdgcn_global_load_async_to_lds_b128(                            \
        (v4i*)(Acopy + (kkv)), (v4i*)((bufp) ? ldsDst1 : ldsDst0), 0, 0)
#else
#define ISSUE_COPY(bufp, kkv)                                                  \
    do {                                                                       \
        float* _d = (bufp) ? ldsDst1 : ldsDst0;                                \
        *(float4*)_d = *(const float4*)(Acopy + (kkv));                        \
    } while (0)
#endif

    v8f acc = {0.f, 0.f, 0.f, 0.f, 0.f, 0.f, 0.f, 0.f};

    // Loop-invariant pieces of the per-lane B offset (all 32-bit).
    const int bOff    = col0 + mn + half * 8 * ldb;   // lane column + half*8 K row
    const int sStride = 32 * ldb;                     // second 16x16x32 sub-step

    const int nIter = K / KSTEP;
    ISSUE_COPY(0, 0);

    for (int it = 0; it < nIter; ++it) {
        const int kk = it * KSTEP;
        __syncthreads();                         // buffer (it+1)&1 free to fill
        if (it + 1 < nIter) ISSUE_COPY((it + 1) & 1, kk + KSTEP);
#if USE_ASYNC_LDS
        if (it + 1 < nIter) __builtin_amdgcn_s_wait_asynccnt(1);
        else                __builtin_amdgcn_s_wait_asynccnt(0);
#endif
        __syncthreads();                         // tile it&1 visible to block

        const float* __restrict__ tb   = &tileA[it & 1][mn * KSTEP];
        const float* __restrict__ Brow = B + (long long)kk * ldb;  // uniform
#pragma unroll
        for (int s = 0; s < 2; ++s) {            // two 16x16x32 WMMAs per tile
            // A fragment: two 16B-aligned contiguous 8-float groups from LDS.
            const float4 a0 = *(const float4*)(tb + s * 32 + half * 8);
            const float4 a1 = *(const float4*)(tb + s * 32 + half * 8 + 4);
            const float4 a2 = *(const float4*)(tb + s * 32 + 16 + half * 8);
            const float4 a3 = *(const float4*)(tb + s * 32 + 16 + half * 8 + 4);
            v16h a;
            a[0]  = (_Float16)a0.x; a[1]  = (_Float16)a0.y;
            a[2]  = (_Float16)a0.z; a[3]  = (_Float16)a0.w;
            a[4]  = (_Float16)a1.x; a[5]  = (_Float16)a1.y;
            a[6]  = (_Float16)a1.z; a[7]  = (_Float16)a1.w;
            a[8]  = (_Float16)a2.x; a[9]  = (_Float16)a2.y;
            a[10] = (_Float16)a2.z; a[11] = (_Float16)a2.w;
            a[12] = (_Float16)a3.x; a[13] = (_Float16)a3.y;
            a[14] = (_Float16)a3.z; a[15] = (_Float16)a3.w;

            // B fragment: j*ldb is loop-invariant (hoisted); 1 add per load.
            const int sb = bOff + s * sStride;
            v16h b;
#pragma unroll
            for (int j = 0; j < 16; ++j) {
                b[j] = (_Float16)Brow[sb + (j < 8 ? j : j + 8) * ldb];
            }
            // (neg_a, A, neg_b, B, c_mod, C, reuse_a, reuse_b)
            acc = __builtin_amdgcn_wmma_f32_16x16x32_f16(
                false, a, false, b, (short)0, acc, false, false);
        }
    }
#undef ISSUE_COPY

    const int col = col0 + mn;
    const float bb = bias ? bias[col] : 0.f;
#pragma unroll
    for (int r = 0; r < 8; ++r) {
        const int row = row0 + r + half * 8;
        if (row < M) {
            float v = acc[r] + bb;
            if (rowAdd) v += rowAdd[(long long)row * ldRowAdd + col];
            C[(long long)row * ldc + col] = v;
        }
    }
}

// ---------------------------------------------------------------------------
// Zero a float region (also zeros the uint max-encoding buffers: bits 0).
// ---------------------------------------------------------------------------
__global__ void zero_f32(float* __restrict__ p, long long n)
{
    long long i = (long long)blockIdx.x * blockDim.x + threadIdx.x;
    if (i < n) p[i] = 0.f;
}

// ---------------------------------------------------------------------------
// Per-edge attention scores: one wave per (edge, head). 512-wide dot product,
// butterfly reduce, scale by p_rel/sqrt(HD), then ordered-uint atomicMax into
// the per-(dst,head) segment max.
// ---------------------------------------------------------------------------
__global__ __launch_bounds__(256)
void edge_scores(const int* __restrict__ src_idx, const int* __restrict__ dst_idx,
                 const float* __restrict__ krel, const float* __restrict__ q,
                 const float* __restrict__ p_rel,
                 float* __restrict__ score, unsigned* __restrict__ maxEnc, int E)
{
    const int gw   = (int)((blockIdx.x * (long long)blockDim.x + threadIdx.x) >> 5);
    const int lane = threadIdx.x & 31;
    if (gw >= E * HEADS) return;
    const int e = gw >> 3;
    const int h = gw & 7;
    const int s = src_idx[e];
    const int d = dst_idx[e];
    const float* __restrict__ kv = krel + (long long)s * D_OUT + h * HD;
    const float* __restrict__ qv = q    + (long long)d * D_OUT + h * HD;

    float sum = 0.f;
#pragma unroll
    for (int j = 0; j < 16; ++j)
        sum += kv[lane + 32 * j] * qv[lane + 32 * j];
#pragma unroll
    for (int off = 16; off > 0; off >>= 1)
        sum += __shfl_xor(sum, off, 32);

    if (lane == 0) {
        const float sc = sum * p_rel[h] * 0.04419417382415922f; // 1/sqrt(512)
        score[(long long)e * HEADS + h] = sc;
        unsigned u = __float_as_uint(sc);
        u = (u & 0x80000000u) ? ~u : (u | 0x80000000u); // total-order encoding
        atomicMax(&maxEnc[d * HEADS + h], u);
    }
}

// ---------------------------------------------------------------------------
// ex = exp(score - segmax); segment-sum via atomicAdd. score overwritten w/ ex.
// ---------------------------------------------------------------------------
__global__ void edge_expsum(const int* __restrict__ dst_idx,
                            float* __restrict__ score,
                            const unsigned* __restrict__ maxEnc,
                            float* __restrict__ sumBuf, int E)
{
    const int i = blockIdx.x * blockDim.x + threadIdx.x;
    if (i >= E * HEADS) return;
    const int e = i >> 3, h = i & 7;
    const int d = dst_idx[e];
    const unsigned u = maxEnc[d * HEADS + h];
    const float m = (u & 0x80000000u) ? __uint_as_float(u ^ 0x80000000u)
                                      : __uint_as_float(~u);
    const float ex = expf(score[i] - m);
    score[i] = ex;
    atomicAdd(&sumBuf[d * HEADS + h], ex);
}

// ---------------------------------------------------------------------------
// Scatter messages: agg[dst] += v_rel[src] * alpha.  One block per edge.
// agg (11 MB total) stays resident in the 192 MB L2 so atomics resolve at L2.
// ---------------------------------------------------------------------------
__global__ __launch_bounds__(256)
void edge_scatter(const int* __restrict__ src_idx, const int* __restrict__ dst_idx,
                  const float* __restrict__ score, const float* __restrict__ sumBuf,
                  const float* __restrict__ vrel, float* __restrict__ agg, int E)
{
    const int e = blockIdx.x;
    const int s = src_idx[e];
    const int d = dst_idx[e];
    for (int i = threadIdx.x; i < D_OUT; i += 256) {
        const int h = i >> 9;
        const float alpha = score[(long long)e * HEADS + h] /
                            (sumBuf[d * HEADS + h] + 1e-16f);
        atomicAdd(&agg[(long long)d * D_OUT + i],
                  vrel[(long long)s * D_OUT + i] * alpha);
    }
}

// ---------------------------------------------------------------------------
// jax.nn.gelu (tanh approximation), in place.
// ---------------------------------------------------------------------------
__global__ void gelu_inplace(float* __restrict__ x, long long n)
{
    long long i = (long long)blockIdx.x * blockDim.x + threadIdx.x;
    if (i < n) {
        const float v = x[i];
        const float c = 0.7978845608028654f * (v + 0.044715f * v * v * v);
        x[i] = 0.5f * v * (1.f + tanhf(c));
    }
}

// ---------------------------------------------------------------------------
// LayerNorm over rows of 4096, in place on d_out. One 256-thread block / row.
// ---------------------------------------------------------------------------
__global__ __launch_bounds__(256)
void layernorm_rows(float* __restrict__ x, const float* __restrict__ g,
                    const float* __restrict__ b)
{
    __shared__ float red[2][8];
    float* __restrict__ xr = x + (long long)blockIdx.x * D_OUT;

    float s = 0.f, s2 = 0.f;
    for (int i = threadIdx.x; i < D_OUT; i += 256) {
        const float v = xr[i];
        s += v; s2 += v * v;
    }
#pragma unroll
    for (int off = 16; off > 0; off >>= 1) {
        s  += __shfl_xor(s, off, 32);
        s2 += __shfl_xor(s2, off, 32);
    }
    const int wave = threadIdx.x >> 5, lane = threadIdx.x & 31;
    if (lane == 0) { red[0][wave] = s; red[1][wave] = s2; }
    __syncthreads();
    if (threadIdx.x == 0) {
        float ts = 0.f, ts2 = 0.f;
        for (int w = 0; w < 8; ++w) { ts += red[0][w]; ts2 += red[1][w]; }
        red[0][0] = ts; red[1][0] = ts2;
    }
    __syncthreads();
    const float mean = red[0][0] * (1.f / D_OUT);
    const float var  = red[1][0] * (1.f / D_OUT) - mean * mean;
    const float inv  = rsqrtf(var + 1e-12f);
    for (int i = threadIdx.x; i < D_OUT; i += 256)
        xr[i] = (xr[i] - mean) * inv * g[i] + b[i];
}

// ---------------------------------------------------------------------------
extern "C" void kernel_launch(void* const* d_in, const int* in_sizes, int n_in,
                              void* d_out, int out_size, void* d_ws, size_t ws_size,
                              hipStream_t stream)
{
    const float* xP    = (const float*)d_in[0];
    const float* xA    = (const float*)d_in[1];
    const int*   ei_pa = (const int*)d_in[2];
    const int*   ei_ap = (const int*)d_in[3];
    const float* kWp   = (const float*)d_in[4];
    const float* kbp   = (const float*)d_in[5];
    const float* qWp   = (const float*)d_in[6];
    const float* qbp   = (const float*)d_in[7];
    const float* vWp   = (const float*)d_in[8];
    const float* vbp   = (const float*)d_in[9];
    const float* outWp = (const float*)d_in[10];
    const float* outbp = (const float*)d_in[11];
    /* d_in[12] = skip_paper (unused: skip reduces to identity) */
    const float* kWa   = (const float*)d_in[13];
    const float* kba   = (const float*)d_in[14];
    const float* qWa   = (const float*)d_in[15];
    const float* qba   = (const float*)d_in[16];
    const float* vWa   = (const float*)d_in[17];
    const float* vba   = (const float*)d_in[18];
    const float* outWa = (const float*)d_in[19];
    const float* outba = (const float*)d_in[20];
    /* d_in[21] = skip_author (unused) */
    const float* arel0 = (const float*)d_in[22];
    const float* mrel0 = (const float*)d_in[23];
    const float* prel0 = (const float*)d_in[24];
    const float* arel1 = (const float*)d_in[25];
    const float* mrel1 = (const float*)d_in[26];
    const float* prel1 = (const float*)d_in[27];
    const float* nodeW = (const float*)d_in[28];
    const float* nodeb = (const float*)d_in[29];
    const float* pos   = (const float*)d_in[30];
    const float* lng   = (const float*)d_in[31];
    const float* lnb   = (const float*)d_in[32];

    const int NPn = in_sizes[0] / 768;   // 400 paper nodes
    const int NAn = in_sizes[1] / 768;   // 300 author nodes
    const int E0  = in_sizes[2] / 2;     // 60000 paper->author edges
    const int E1  = in_sizes[3] / 2;     // 60000 author->paper edges

    // ---- workspace layout (floats); total ~21.5M floats (~86 MB) ----------
    float* W = (float*)d_ws;
    float* kP    = W;                 W += (long long)NPn * D_OUT;
    float* qP    = W;                 W += (long long)NPn * D_OUT;
    float* vP    = W;                 W += (long long)NPn * D_OUT;
    float* kA    = W;                 W += (long long)NAn * D_OUT;
    float* qA    = W;                 W += (long long)NAn * D_OUT;
    float* vA    = W;                 W += (long long)NAn * D_OUT;
    float* krel0 = W;                 W += (long long)NPn * D_OUT;
    float* vrel0 = W;                 W += (long long)NPn * D_OUT;
    float* krel1 = W;                 W += (long long)NAn * D_OUT;
    float* vrel1 = W;                 W += (long long)NAn * D_OUT;
    // ---- zeroed-every-call region starts here ----
    float*    zeroBase = W;
    float*    aggP = W;               W += (long long)NPn * D_OUT;
    float*    aggA = W;               W += (long long)NAn * D_OUT;
    float*    sum0 = W;               W += 512 * HEADS;   // dst=author
    float*    sum1 = W;               W += 512 * HEADS;   // dst=paper
    unsigned* max0 = (unsigned*)W;    W += 512 * HEADS;
    unsigned* max1 = (unsigned*)W;    W += 512 * HEADS;
    const long long zeroCount = W - zeroBase;
    // ---- end zeroed region ----
    float* score0 = W;                W += (long long)E0 * HEADS;
    float* score1 = W;                W += (long long)E1 * HEADS;
    float* hP     = W;                W += (long long)NPn * D_OUT;
    float* hA     = W;                W += (long long)NPn * D_OUT;
    (void)ws_size; (void)n_in; (void)out_size;

    float* outP = (float*)d_out;                       // [NPn, 4096]
    float* outA = (float*)d_out + (long long)NPn * D_OUT;

    const dim3 blk(256);
    auto gemm = [&](const float* A, int lda, long long sA,
                    const float* B, int ldb, long long sB,
                    float* C, int ldc, long long sC,
                    const float* bias, const float* rowAdd, int ldR,
                    int M, int N, int K, int batch) {
        dim3 g((N + 127) / 128, (M + 15) / 16, batch);
        gemm_wmma_f16<<<g, blk, 0, stream>>>(A, lda, sA, B, ldb, sB,
                                             C, ldc, sC, bias, rowAdd, ldR,
                                             M, N, K);
    };

    // 0) zero accumulators (agg + segment sum/max)
    zero_f32<<<dim3((unsigned)((zeroCount + 255) / 256)), blk, 0, stream>>>(
        zeroBase, zeroCount);

    // 1) K/Q/V projections: [N,768] @ [768,4096] + bias
    gemm(xP, 768, 0, kWp, D_OUT, 0, kP, D_OUT, 0, kbp, nullptr, 0, NPn, D_OUT, 768, 1);
    gemm(xP, 768, 0, qWp, D_OUT, 0, qP, D_OUT, 0, qbp, nullptr, 0, NPn, D_OUT, 768, 1);
    gemm(xP, 768, 0, vWp, D_OUT, 0, vP, D_OUT, 0, vbp, nullptr, 0, NPn, D_OUT, 768, 1);
    gemm(xA, 768, 0, kWa, D_OUT, 0, kA, D_OUT, 0, kba, nullptr, 0, NAn, D_OUT, 768, 1);
    gemm(xA, 768, 0, qWa, D_OUT, 0, qA, D_OUT, 0, qba, nullptr, 0, NAn, D_OUT, 768, 1);
    gemm(xA, 768, 0, vWa, D_OUT, 0, vA, D_OUT, 0, vba, nullptr, 0, NAn, D_OUT, 768, 1);

    // 2) relation transforms (per node, NOT per edge): 8-head batched GEMMs
    //    k_rel[n,h,:] = k[n,h,:] @ a_rel[h]   ([N,512] @ [512,512] per head)
    gemm(kP, D_OUT, HD, arel0, HD, (long long)HD * HD, krel0, D_OUT, HD,
         nullptr, nullptr, 0, NPn, HD, HD, HEADS);
    gemm(vP, D_OUT, HD, mrel0, HD, (long long)HD * HD, vrel0, D_OUT, HD,
         nullptr, nullptr, 0, NPn, HD, HD, HEADS);
    gemm(kA, D_OUT, HD, arel1, HD, (long long)HD * HD, krel1, D_OUT, HD,
         nullptr, nullptr, 0, NAn, HD, HD, HEADS);
    gemm(vA, D_OUT, HD, mrel1, HD, (long long)HD * HD, vrel1, D_OUT, HD,
         nullptr, nullptr, 0, NAn, HD, HD, HEADS);

    // 3) edge type 0: paper -> author (dst = author)
    {
        const int* src = ei_pa;
        const int* dst = ei_pa + E0;
        edge_scores<<<dim3((unsigned)(((long long)E0 * HEADS * 32 + 255) / 256)),
                      blk, 0, stream>>>(src, dst, krel0, qA, prel0, score0, max0, E0);
        edge_expsum<<<dim3((unsigned)((E0 * HEADS + 255) / 256)), blk, 0, stream>>>(
            dst, score0, max0, sum0, E0);
        edge_scatter<<<dim3((unsigned)E0), blk, 0, stream>>>(
            src, dst, score0, sum0, vrel0, aggA, E0);
    }
    // 4) edge type 1: author -> paper (dst = paper)
    {
        const int* src = ei_ap;
        const int* dst = ei_ap + E1;
        edge_scores<<<dim3((unsigned)(((long long)E1 * HEADS * 32 + 255) / 256)),
                      blk, 0, stream>>>(src, dst, krel1, qP, prel1, score1, max1, E1);
        edge_expsum<<<dim3((unsigned)((E1 * HEADS + 255) / 256)), blk, 0, stream>>>(
            dst, score1, max1, sum1, E1);
        edge_scatter<<<dim3((unsigned)E1), blk, 0, stream>>>(
            src, dst, score1, sum1, vrel1, aggP, E1);
    }

    // 5) gelu on agg (aggP/aggA are contiguous -> one launch)
    {
        const long long n = (long long)(NPn + NAn) * D_OUT;
        gelu_inplace<<<dim3((unsigned)((n + 255) / 256)), blk, 0, stream>>>(aggP, n);
    }

    // 6) out projection: h = gelu(agg) @ outW + outb  (skip-gate is identity)
    gemm(aggP, D_OUT, 0, outWp, D_OUT, 0, hP, D_OUT, 0, outbp, nullptr, 0,
         NPn, D_OUT, D_OUT, 1);
    gemm(aggA, D_OUT, 0, outWa, D_OUT, 0, hA, D_OUT, 0, outba, nullptr, 0,
         NAn, D_OUT, D_OUT, 1);

    // 7) node projection + positional embedding (row-wise add), into d_out
    gemm(hP, D_OUT, 0, nodeW, D_OUT, 0, outP, D_OUT, 0, nodeb, pos, D_OUT,
         NPn, D_OUT, D_OUT, 1);
    gemm(hA, D_OUT, 0, nodeW, D_OUT, 0, outA, D_OUT, 0, nodeb, pos, D_OUT,
         NAn, D_OUT, D_OUT, 1);

    // 8) layernorm in place on d_out (700 rows)
    layernorm_rows<<<dim3((unsigned)(NPn + NAn)), blk, 0, stream>>>(outP, lng, lnb);
}
